// Network_2147483648178
// MI455X (gfx1250) — matrix operand order repaired and checked by reading, not compiled, not run on hardware
//
#include <hip/hip_runtime.h>
#include <hip/hip_bf16.h>

// ---------------- model dims ----------------
#define Bn 4
#define Ln 2048
#define Mtok (Bn*Ln)          // 8192
#define IN_DIM 64
#define OUT_DIM 2
#define DM 768
#define DI 1536
#define DSTATE 16
#define DTRANK 48
#define DTPAD 64
#define DCONV 4
#define XPN (DTRANK + 2*DSTATE)  // 80 (true width)
#define XPNP 128                 // padded to a multiple of 64 for guard-free WMMA

typedef __attribute__((ext_vector_type(16))) __bf16 v16bf;
typedef __attribute__((ext_vector_type(8)))  float  v8f;

union Frag16 { v16bf v; uint4 q[2]; unsigned short u[16]; };

__device__ __forceinline__ unsigned short f2bf(float f) {
  unsigned int u = __float_as_uint(f);
  unsigned int r = u + 0x7FFFu + ((u >> 16) & 1u);
  return (unsigned short)(r >> 16);
}
__device__ __forceinline__ float bf2f(unsigned short s) {
  return __uint_as_float(((unsigned int)s) << 16);
}
__device__ __forceinline__ float silu(float x) {
  return x / (1.0f + __expf(-x));
}

// ---------------- generic f32 -> bf16 convert ----------------
__global__ void k_cvt_bf16(const float* __restrict__ in, unsigned short* __restrict__ out, int n) {
  int i = blockIdx.x * blockDim.x + threadIdx.x;
  if (i < n) out[i] = f2bf(in[i]);
}

// x_proj_w (80 x DI) -> zero-padded bf16 (128 x DI)
__global__ void k_xpw_pad(const float* __restrict__ w, unsigned short* __restrict__ wp) {
  int i = blockIdx.x * blockDim.x + threadIdx.x;
  if (i >= XPNP * DI) return;
  int n = i / DI, k = i % DI;
  wp[i] = (n < XPN) ? f2bf(w[n * DI + k]) : (unsigned short)0;
}

// dt_proj_w (DI x 48) -> padded bf16 (DI x 64)
__global__ void k_dtw_pad(const float* __restrict__ w, unsigned short* __restrict__ wp) {
  int i = blockIdx.x * blockDim.x + threadIdx.x;
  if (i >= DI * DTPAD) return;
  int n = i >> 6, k = i & 63;
  wp[i] = (k < DTRANK) ? f2bf(w[n * DTRANK + k]) : (unsigned short)0;
}

// dbl[:, :48] (stride XPNP) -> padded bf16 activation (M x 64)
__global__ void k_dr_pad(const float* __restrict__ dbl, unsigned short* __restrict__ drp) {
  int i = blockIdx.x * blockDim.x + threadIdx.x;
  if (i >= Mtok * DTPAD) return;
  int m = i >> 6, k = i & 63;
  drp[i] = (k < DTRANK) ? f2bf(dbl[(size_t)m * XPNP + k]) : (unsigned short)0;
}

// ---------------- WMMA bf16 GEMM:  C[M,N] (f32) = A[M,K](bf16) * B[N,K](bf16)^T (+bias)(+=C)
// Guard-free: requires M % 128 == 0, N % 64 == 0, K % 32 == 0 (callers pad).
// wave: 32 rows x 64 cols (8 WMMAs / K-step); block: 128 thr = 4 waves -> 128x64 tile.
__global__ __launch_bounds__(128)
void k_gemm_bf16(float* __restrict__ C,
                 const unsigned short* __restrict__ A,
                 const unsigned short* __restrict__ Bw,
                 const float* __restrict__ bias,
                 int M, int N, int K, int accumulate) {
  const int lane = threadIdx.x & 31;
  const int wave = threadIdx.x >> 5;
  const int half = lane >> 4;
  const int lrow = lane & 15;
  const int m0 = blockIdx.y * 128 + wave * 32;
  const int n0 = blockIdx.x * 64;

  v8f acc[8];
#pragma unroll
  for (int i = 0; i < 8; ++i) acc[i] = v8f{};

  // A 16x32 bf16 fragment: lane m holds K = 8*half + [0..7] and 16 + 8*half + [0..7]
  const unsigned short* Arow0 = A + (size_t)(m0 + lrow) * K;
  const unsigned short* Arow1 = A + (size_t)(m0 + 16 + lrow) * K;
  // B 32x16 bf16 fragment: lane n holds K = 8*half + [0..7] and 16 + 8*half + [0..7] of row n of the (N,K) weight
  const unsigned short* Bbase = Bw + (size_t)(n0 + lrow) * K + 8 * half;

  for (int k = 0; k < K; k += 32) {
    Frag16 a0, a1;
    a0.q[0] = *(const uint4*)(Arow0 + k + 8 * half);
    a0.q[1] = *(const uint4*)(Arow0 + k + 16 + 8 * half);
    a1.q[0] = *(const uint4*)(Arow1 + k + 8 * half);
    a1.q[1] = *(const uint4*)(Arow1 + k + 16 + 8 * half);
#pragma unroll
    for (int t = 0; t < 4; ++t) {
      Frag16 bfg;
      const unsigned short* Brow = Bbase + (size_t)(t * 16) * K + k;
      bfg.q[0] = *(const uint4*)(Brow);
      bfg.q[1] = *(const uint4*)(Brow + 16);
      acc[t] = __builtin_amdgcn_wmma_f32_16x16x32_bf16(
          false, a0.v, false, bfg.v, (short)0, acc[t], false, false);
      acc[4 + t] = __builtin_amdgcn_wmma_f32_16x16x32_bf16(
          false, a1.v, false, bfg.v, (short)0, acc[4 + t], false, false);
    }
  }

#pragma unroll
  for (int t = 0; t < 4; ++t) {
    const int n = n0 + t * 16 + lrow;
    const float bv = bias ? bias[n] : 0.0f;
#pragma unroll
    for (int i = 0; i < 8; ++i) {
      const int m = m0 + i + 8 * half;           // C/D layout: VGPR i -> row i + 8*half
      const size_t idx0 = (size_t)m * N + n;
      float v0 = acc[t][i] + bv;
      if (accumulate) v0 += C[idx0];
      C[idx0] = v0;
      const size_t idx1 = idx0 + (size_t)16 * N;
      float v1 = acc[4 + t][i] + bv;
      if (accumulate) v1 += C[idx1];
      C[idx1] = v1;
    }
  }
}

// ---------------- rmsnorm: per token over D, write bf16 ----------------
__global__ __launch_bounds__(256)
void k_rmsnorm(const float* __restrict__ x, const float* __restrict__ w,
               unsigned short* __restrict__ out, int D) {
  const int m = blockIdx.x;
  const float* xr = x + (size_t)m * D;
  float ss = 0.0f;
  for (int i = threadIdx.x; i < D; i += 256) { float v = xr[i]; ss += v * v; }
  for (int o = 16; o > 0; o >>= 1) ss += __shfl_down(ss, o, 32);
  __shared__ float red[8];
  __shared__ float s_r;
  if ((threadIdx.x & 31) == 0) red[threadIdx.x >> 5] = ss;
  __syncthreads();
  if (threadIdx.x == 0) {
    float t = 0.0f;
#pragma unroll
    for (int i = 0; i < 8; ++i) t += red[i];
    s_r = rsqrtf(t / (float)D + 1e-5f);
  }
  __syncthreads();
  const float r = s_r;
  for (int i = threadIdx.x; i < D; i += 256)
    out[(size_t)m * D + i] = f2bf(xr[i] * r * w[i]);
}

// ---------------- causal depthwise conv (k=4) + bias + SiLU ----------------
__global__ void k_conv_silu(const float* __restrict__ xr,      // (M, 2*DI), cols [0,DI) = xs
                            const float* __restrict__ cw,      // (DI, 4)
                            const float* __restrict__ cb,      // (DI)
                            float* __restrict__ xc, unsigned short* __restrict__ xc_bf) {
  int idx = blockIdx.x * blockDim.x + threadIdx.x;
  if (idx >= Mtok * DI) return;
  const int d = idx % DI;
  const int m = idx / DI;
  const int t = m % Ln;
  const int b0 = m - t;  // b*Ln
  float acc = cb[d];
#pragma unroll
  for (int j = 0; j < DCONV; ++j) {
    const int tt = t - (DCONV - 1) + j;
    if (tt >= 0) acc += xr[(size_t)(b0 + tt) * (2 * DI) + d] * cw[d * DCONV + j];
  }
  const float s = silu(acc);
  xc[idx] = s;
  xc_bf[idx] = f2bf(s);
}

// ---------------- softplus in-place ----------------
__global__ void k_softplus(float* __restrict__ x, int n) {
  int i = blockIdx.x * blockDim.x + threadIdx.x;
  if (i >= n) return;
  const float v = x[i];
  x[i] = (v > 20.0f) ? v : log1pf(__expf(v));
}

// ---------------- selective scan: thread per (b,d), serial over L ----------------
__global__ __launch_bounds__(256)
void k_scan(const float* __restrict__ delta, const float* __restrict__ dbl,
            const float* __restrict__ xc, const float* __restrict__ A_log,
            float* __restrict__ ys) {
  const int d = blockIdx.x * blockDim.x + threadIdx.x;
  const int b = blockIdx.y;
  if (d >= DI) return;
  float Ar[DSTATE], s[DSTATE];
#pragma unroll
  for (int n = 0; n < DSTATE; ++n) {
    Ar[n] = -__expf(A_log[d * DSTATE + n]);
    s[n] = 0.0f;
  }
  const size_t base = (size_t)b * Ln;
  for (int t = 0; t < Ln; ++t) {
    const size_t m = base + t;
    const float dt = delta[m * DI + d];
    const float u = xc[m * DI + d];
    const float du = dt * u;
    const float* Bt = dbl + m * XPNP + DTRANK;
    const float* Ct = dbl + m * XPNP + DTRANK + DSTATE;
    float y = 0.0f;
#pragma unroll
    for (int n = 0; n < DSTATE; ++n) {
      const float dA = __expf(dt * Ar[n]);
      s[n] = dA * s[n] + du * Bt[n];
      y += s[n] * Ct[n];
    }
    ys[m * DI + d] = y;
  }
}

// ---------------- y = (ys + xc*D) * silu(res) -> bf16 ----------------
__global__ void k_gate(const float* __restrict__ ys, const float* __restrict__ xc,
                       const float* __restrict__ xr, const float* __restrict__ Dp,
                       unsigned short* __restrict__ y_bf) {
  int idx = blockIdx.x * blockDim.x + threadIdx.x;
  if (idx >= Mtok * DI) return;
  const int d = idx % DI;
  const int m = idx / DI;
  const float y = ys[idx] + xc[idx] * Dp[d];
  const float r = xr[(size_t)m * (2 * DI) + DI + d];
  y_bf[idx] = f2bf(y * silu(r));
}

// ---------------- logits (N=2, scalar dot) ----------------
__global__ void k_logits(const unsigned short* __restrict__ hn_bf,
                         const unsigned short* __restrict__ w2_bf,
                         const float* __restrict__ b2, float* __restrict__ logits) {
  int idx = blockIdx.x * blockDim.x + threadIdx.x;
  if (idx >= Mtok * OUT_DIM) return;
  const int o = idx & 1;
  const int m = idx >> 1;
  float acc = b2[o];
  const unsigned short* hr = hn_bf + (size_t)m * DM;
  const unsigned short* wr = w2_bf + (size_t)o * DM;
  for (int k = 0; k < DM; ++k) acc += bf2f(hr[k]) * bf2f(wr[k]);
  logits[idx] = acc;
}

// ---------------- softmax over L per (b, o) ----------------
__global__ __launch_bounds__(256)
void k_softmax_L(const float* __restrict__ logits, float* __restrict__ out) {
  const int b = blockIdx.x >> 1;
  const int o = blockIdx.x & 1;
  const float* lp = logits + (size_t)b * Ln * OUT_DIM + o;
  __shared__ float sm[256];
  float mx = -1e30f;
  for (int t = threadIdx.x; t < Ln; t += 256) mx = fmaxf(mx, lp[(size_t)t * OUT_DIM]);
  sm[threadIdx.x] = mx;
  __syncthreads();
  for (int s = 128; s > 0; s >>= 1) {
    if (threadIdx.x < s) sm[threadIdx.x] = fmaxf(sm[threadIdx.x], sm[threadIdx.x + s]);
    __syncthreads();
  }
  const float M = sm[0];
  __syncthreads();
  float sum = 0.0f;
  for (int t = threadIdx.x; t < Ln; t += 256) sum += __expf(lp[(size_t)t * OUT_DIM] - M);
  sm[threadIdx.x] = sum;
  __syncthreads();
  for (int s = 128; s > 0; s >>= 1) {
    if (threadIdx.x < s) sm[threadIdx.x] += sm[threadIdx.x + s];
    __syncthreads();
  }
  const float inv = 1.0f / sm[0];
  for (int t = threadIdx.x; t < Ln; t += 256)
    out[((size_t)b * Ln + t) * OUT_DIM + o] = __expf(lp[(size_t)t * OUT_DIM] - M) * inv;
}

// =====================================================================
extern "C" void kernel_launch(void* const* d_in, const int* in_sizes, int n_in,
                              void* d_out, int out_size, void* d_ws, size_t ws_size,
                              hipStream_t stream) {
  const float* x        = (const float*)d_in[0];   // (4,2048,64)
  const float* W1       = (const float*)d_in[1];   // (768,64)
  const float* b1       = (const float*)d_in[2];   // (768)
  const float* norm_w   = (const float*)d_in[3];   // (4,768)
  const float* in_w     = (const float*)d_in[4];   // (4,3072,768)
  const float* conv_w   = (const float*)d_in[5];   // (4,1536,4)
  const float* conv_b   = (const float*)d_in[6];   // (4,1536)
  const float* xp_w     = (const float*)d_in[7];   // (4,80,1536)
  const float* dt_w     = (const float*)d_in[8];   // (4,1536,48)
  const float* dt_b     = (const float*)d_in[9];   // (4,1536)
  const float* A_log    = (const float*)d_in[10];  // (4,1536,16)
  const float* Dp       = (const float*)d_in[11];  // (4,1536)
  const float* out_w    = (const float*)d_in[12];  // (4,768,1536)
  const float* normf_w  = (const float*)d_in[13];  // (768)
  const float* W2       = (const float*)d_in[14];  // (2,768)
  const float* b2       = (const float*)d_in[15];  // (2)
  float* outp = (float*)d_out;

  // ---- bump allocator over d_ws ----
  char* ws = (char*)d_ws;
  size_t off = 0;
  auto alloc = [&](size_t bytes) -> void* {
    off = (off + 255) & ~(size_t)255;
    void* p = ws + off;
    off += bytes;
    return p;
  };
  float* h      = (float*)alloc((size_t)Mtok * DM * 4);
  float* xr     = (float*)alloc((size_t)Mtok * 2 * DI * 4);
  float* xc     = (float*)alloc((size_t)Mtok * DI * 4);
  float* delta  = (float*)alloc((size_t)Mtok * DI * 4);
  float* ysb    = (float*)alloc((size_t)Mtok * DI * 4);
  float* dbl    = (float*)alloc((size_t)Mtok * XPNP * 4);
  float* logits = (float*)alloc((size_t)Mtok * OUT_DIM * 4);
  unsigned short* xn_bf  = (unsigned short*)alloc((size_t)Mtok * DM * 2);
  unsigned short* xc_bf  = (unsigned short*)alloc((size_t)Mtok * DI * 2);
  unsigned short* drp_bf = (unsigned short*)alloc((size_t)Mtok * DTPAD * 2);
  unsigned short* y_bf   = (unsigned short*)alloc((size_t)Mtok * DI * 2);
  unsigned short* hn_bf  = (unsigned short*)alloc((size_t)Mtok * DM * 2);
  unsigned short* xin_bf = (unsigned short*)alloc((size_t)Mtok * IN_DIM * 2);
  unsigned short* w1_bf  = (unsigned short*)alloc((size_t)DM * IN_DIM * 2);
  unsigned short* inw_bf = (unsigned short*)alloc((size_t)2 * DI * DM * 2);
  unsigned short* xpw_bf = (unsigned short*)alloc((size_t)XPNP * DI * 2);
  unsigned short* dtw_bf = (unsigned short*)alloc((size_t)DI * DTPAD * 2);
  unsigned short* ow_bf  = (unsigned short*)alloc((size_t)DM * DI * 2);
  unsigned short* w2_bf  = (unsigned short*)alloc((size_t)OUT_DIM * DM * 2);

  auto cdiv = [](int a, int b) { return (a + b - 1) / b; };

  // ---- embed: h = x @ W1^T + b1 ----
  k_cvt_bf16<<<cdiv(Mtok * IN_DIM, 256), 256, 0, stream>>>(x, xin_bf, Mtok * IN_DIM);
  k_cvt_bf16<<<cdiv(DM * IN_DIM, 256), 256, 0, stream>>>(W1, w1_bf, DM * IN_DIM);
  k_gemm_bf16<<<dim3(DM / 64, Mtok / 128), 128, 0, stream>>>(
      h, xin_bf, w1_bf, b1, Mtok, DM, IN_DIM, 0);

  // ---- layers ----
  for (int l = 0; l < 4; ++l) {
    // rmsnorm -> bf16
    k_rmsnorm<<<Mtok, 256, 0, stream>>>(h, norm_w + (size_t)l * DM, xn_bf, DM);
    // in_proj
    k_cvt_bf16<<<cdiv(2 * DI * DM, 256), 256, 0, stream>>>(
        in_w + (size_t)l * 2 * DI * DM, inw_bf, 2 * DI * DM);
    k_gemm_bf16<<<dim3((2 * DI) / 64, Mtok / 128), 128, 0, stream>>>(
        xr, xn_bf, inw_bf, nullptr, Mtok, 2 * DI, DM, 0);
    // conv + silu
    k_conv_silu<<<cdiv(Mtok * DI, 256), 256, 0, stream>>>(
        xr, conv_w + (size_t)l * DI * DCONV, conv_b + (size_t)l * DI, xc, xc_bf);
    // x_proj -> dbl (N padded 80 -> 128, guard-free)
    k_xpw_pad<<<cdiv(XPNP * DI, 256), 256, 0, stream>>>(
        xp_w + (size_t)l * XPN * DI, xpw_bf);
    k_gemm_bf16<<<dim3(XPNP / 64, Mtok / 128), 128, 0, stream>>>(
        dbl, xc_bf, xpw_bf, nullptr, Mtok, XPNP, DI, 0);
    // dt_proj: pad K 48->64, GEMM, softplus
    k_dr_pad<<<cdiv(Mtok * DTPAD, 256), 256, 0, stream>>>(dbl, drp_bf);
    k_dtw_pad<<<cdiv(DI * DTPAD, 256), 256, 0, stream>>>(
        dt_w + (size_t)l * DI * DTRANK, dtw_bf);
    k_gemm_bf16<<<dim3(DI / 64, Mtok / 128), 128, 0, stream>>>(
        delta, drp_bf, dtw_bf, dt_b + (size_t)l * DI, Mtok, DI, DTPAD, 0);
    k_softplus<<<cdiv(Mtok * DI, 256), 256, 0, stream>>>(delta, Mtok * DI);
    // selective scan
    k_scan<<<dim3(DI / 256, Bn), 256, 0, stream>>>(
        delta, dbl, xc, A_log + (size_t)l * DI * DSTATE, ysb);
    // gate
    k_gate<<<cdiv(Mtok * DI, 256), 256, 0, stream>>>(
        ysb, xc, xr, Dp + (size_t)l * DI, y_bf);
    // out_proj with residual accumulate into h
    k_cvt_bf16<<<cdiv(DM * DI, 256), 256, 0, stream>>>(
        out_w + (size_t)l * DM * DI, ow_bf, DM * DI);
    k_gemm_bf16<<<dim3(DM / 64, Mtok / 128), 128, 0, stream>>>(
        h, y_bf, ow_bf, nullptr, Mtok, DM, DI, 1);
  }

  // ---- head ----
  k_rmsnorm<<<Mtok, 256, 0, stream>>>(h, normf_w, hn_bf, DM);
  k_cvt_bf16<<<cdiv(OUT_DIM * DM, 256), 256, 0, stream>>>(W2, w2_bf, OUT_DIM * DM);
  k_logits<<<cdiv(Mtok * OUT_DIM, 256), 256, 0, stream>>>(hn_bf, w2_bf, b2, logits);
  k_softmax_L<<<Bn * OUT_DIM, 256, 0, stream>>>(logits, outp);
}
